// ConvNet_BiLSTM_58256936403653
// MI455X (gfx1250) — compile-verified, matrix operand order checked by reading, not run on hardware
//
#include <hip/hip_runtime.h>
#include <hip/hip_bf16.h>
#include <math.h>

typedef _Float16 f16;
typedef __attribute__((ext_vector_type(16))) _Float16 v16h;
typedef __attribute__((ext_vector_type(8)))  float    v8f;

#define WMMA_F16(a, b, c) \
  __builtin_amdgcn_wmma_f32_16x16x32_f16(false, (a), false, (b), (short)0, (c), false, false)

// ---------------------------------------------------------------------------
// WMMA 16x16x32 f16 fragment geometry (CDNA5 ISA 7.12.2, wave32):
//   lane l in [0,16):  M/N = l,     holds K 0..7 (e 0..7)  and 16..23 (e 8..15)
//   lane l in [16,32): M/N = l-16,  holds K 8..15          and 24..31
// We store tiles "fragment-major": tile[lane][e], so a fragment load is one
// contiguous 32-byte read per lane.
// frag_off(mn, kk): flat offset of element (row mn, k kk) inside a 512-elem tile
// ---------------------------------------------------------------------------
__device__ __forceinline__ int frag_off(int mn, int kk) {
  return ((((kk >> 3) & 1) * 16 + mn) << 4) + ((kk >> 4) << 3) + (kk & 7);
}

__device__ __forceinline__ v16h frag_ld(const f16* __restrict__ p) {
  return *reinterpret_cast<const v16h*>(p + ((threadIdx.x & 31) << 4));
}

// Swizzled read from a row-major [16][ld] f16 LDS tile (used by the async-copy path)
__device__ __forceinline__ v16h frag_a_row(const f16* __restrict__ p, int ld) {
  int l = threadIdx.x & 31;
  int half = l >> 4, m = l & 15;
  const f16* row = p + m * ld;
  v16h a;
#pragma unroll
  for (int e = 0; e < 16; ++e) {
    int k = ((e >> 3) << 4) + (half << 3) + (e & 7);
    a[e] = row[k];
  }
  return a;
}

// ---------------------------------------------------------------------------
// Weight prep: f32 -> f16, transpose/pad/merge, emitted in fragment-major tiles
// ---------------------------------------------------------------------------
__global__ void prep_convw(const float* __restrict__ cw, f16* __restrict__ wp) {
  int i = blockIdx.x * 256 + threadIdx.x;            // 32*256; k = t*4+ic (pad 32)
  int k = i >> 8, c = i & 255;
  float v = 0.f;
  if (k < 28) { int ic = k & 3, t = k >> 2; v = cw[c * 28 + ic * 7 + t]; }
  int nt = c >> 4, nn = c & 15;
  wp[nt * 512 + frag_off(nn, k)] = (f16)v;           // [16 ntiles][512]
}

__global__ void prep_wih(const float* __restrict__ wf, const float* __restrict__ wb,
                         f16* __restrict__ wp) {
  int i = blockIdx.x * 256 + threadIdx.x;            // 256*1024
  int k = i >> 10, n = i & 1023;
  float v = (n < 512) ? wf[n * 256 + k] : wb[(n - 512) * 256 + k];
  int ks = k >> 5, kk = k & 31, nt = n >> 4, nn = n & 15;
  wp[(ks * 64 + nt) * 512 + frag_off(nn, kk)] = (f16)v;   // [8][64][512]
}

__global__ void prep_whh(const float* __restrict__ wf, const float* __restrict__ wb,
                         f16* __restrict__ wp) {
  int i = blockIdx.x * 256 + threadIdx.x;            // 2*128*512
  int d = i >> 16, r = i & 65535;
  int k = r >> 9, n = r & 511;
  const float* w = d ? wb : wf;
  float v = w[n * 128 + k];
  int ks = k >> 5, kk = k & 31, nt = n >> 4, nn = n & 15;
  wp[d * 65536 + (ks * 32 + nt) * 512 + frag_off(nn, kk)] = (f16)v; // [2][4][32][512]
}

__global__ void prep_bias(const float* __restrict__ bihf, const float* __restrict__ bhhf,
                          const float* __restrict__ bihb, const float* __restrict__ bhhb,
                          float* __restrict__ bs) {
  int i = blockIdx.x * 256 + threadIdx.x;            // 1024
  int d = i >> 9, n = i & 511;
  bs[i] = d ? (bihb[n] + bhhb[n]) : (bihf[n] + bhhf[n]);
}

// RepBlock: fold conv3x3 (only dw=1 column survives on W=1 spatial) + conv1x1
__global__ void prep_repw(const float* __restrict__ w3, const float* __restrict__ w1,
                          const float* __restrict__ b3, const float* __restrict__ b1,
                          f16* __restrict__ wr, float* __restrict__ rb) {
  int i = blockIdx.x * 256 + threadIdx.x;            // 768*256
  int k = i >> 8, o = i & 255;
  int wi = k >> 8, ic = k & 255;
  float v = w3[o * 2304 + ic * 9 + wi * 3 + 1];
  if (wi == 1) v += w1[o * 256 + ic];
  int ks = k >> 5, kk = k & 31, nt = o >> 4, nn = o & 15;
  wr[(ks * 16 + nt) * 512 + frag_off(nn, kk)] = (f16)v;   // [24][16][512]
  if (i < 256) rb[i] = b3[i] + b1[i];
}

// Position-linear weights [SCNT][KK][6] f32 -> fragment tiles [SCNT][KK/32][512] f16
__global__ void prep_plw(const float* __restrict__ W, f16* __restrict__ dst,
                         int KK, int total) {
  int i = blockIdx.x * 256 + threadIdx.x;            // SCNT*KK*16
  if (i >= total) return;
  int kk16 = KK * 16;
  int s = i / kk16, r = i - s * kk16;
  int k = r >> 4, n = r & 15;
  float v = (n < 6) ? W[((size_t)s * KK + k) * 6 + n] : 0.f;
  int ks = k >> 5, kl = k & 31;
  dst[((size_t)s * (KK >> 5) + ks) * 512 + frag_off(n, kl)] = (f16)v;
}

__global__ void zero_state(f16* __restrict__ hp, float* __restrict__ cb) {
  int i = blockIdx.x * 256 + threadIdx.x;            // 2*256*128
  hp[i] = (f16)0.f;
  cb[i] = 0.f;
}

// ---------------------------------------------------------------------------
// K0: Conv1d(4->256,k=7,pad=3)+ReLU, K=28 padded into one K=32 WMMA.
// grid (s=256, btile=16), block 128 (4 waves x 4 channel-tiles).
// ---------------------------------------------------------------------------
__global__ void conv_gemm(const float* __restrict__ x, const f16* __restrict__ wp,
                          const float* __restrict__ cb, f16* __restrict__ h16) {
  int s = blockIdx.x, bt = blockIdx.y;
  __shared__ alignas(32) f16 As[512];
  int tid = threadIdx.x;
  for (int i = tid; i < 512; i += 128) {
    int m = i >> 5, k = i & 31;
    float v = 0.f;
    if (k < 28) {
      int ic = k & 3, t = k >> 2, row = s + t - 3;
      if (row >= 0 && row < 256) v = x[((bt * 16 + m) * 256 + row) * 4 + ic];
    }
    As[frag_off(m, k)] = (f16)v;
  }
  __syncthreads();
  int wv = tid >> 5, l = tid & 31, n = l & 15, mh = (l >> 4) * 8;
  v16h a = frag_ld(As);
#pragma unroll
  for (int q = 0; q < 4; ++q) {
    int nt = wv * 4 + q;
    v16h b = frag_ld(wp + nt * 512);
    v8f acc = {};
    acc = WMMA_F16(a, b, acc);
    int c = nt * 16 + n;
    float bias = cb[c];
#pragma unroll
    for (int j = 0; j < 8; ++j) {
      float v = acc[j] + bias;
      v = v > 0.f ? v : 0.f;
      h16[(size_t)((bt * 16 + mh + j) * 256 + s) * 256 + c] = (f16)v;
    }
  }
}

// ---------------------------------------------------------------------------
// K1: pl3 — per-position GEMM M=256,K=768,N=6(pad16). grid (256,16), block 32.
// ---------------------------------------------------------------------------
__global__ void pl3_gemm(const f16* __restrict__ h16, const f16* __restrict__ Wp,
                         const float* __restrict__ bias, float* __restrict__ x3) {
  int s = blockIdx.x, bt = blockIdx.y;
  __shared__ alignas(32) f16 As[24 * 512];           // 24 KB
  int tid = threadIdx.x;                             // 32
  for (int i = tid; i < 16 * 768; i += 32) {
    int m = i / 768, k = i % 768;
    int wi = k >> 8, f = k & 255;
    int row = s + wi - 1;
    f16 v = (f16)0.f;
    if (row >= 0 && row < 256)
      v = h16[(size_t)((bt * 16 + m) * 256 + row) * 256 + f];
    As[(k >> 5) * 512 + frag_off(m, k & 31)] = v;
  }
  __syncthreads();
  const f16* Ws = Wp + (size_t)s * 24 * 512;
  v8f acc = {};
#pragma unroll 4
  for (int ks = 0; ks < 24; ++ks) {
    v16h a = frag_ld(As + ks * 512);
    v16h b = frag_ld(Ws + ks * 512);
    acc = WMMA_F16(a, b, acc);
  }
  int l = tid & 31, n = l & 15, mh = (l >> 4) * 8;
  if (n < 6) {
    float bb = bias[s * 6 + n];
#pragma unroll
    for (int j = 0; j < 8; ++j) {
      int b = bt * 16 + mh + j;
      float v = acc[j] + bb;
      v = v > 0.f ? v : 0.f;
      x3[((size_t)b * 256 + s) * 6 + n] = v;
    }
  }
}

// ---------------------------------------------------------------------------
// K2/K3: pl5/pl7 for s<16 only (windows for s>=16 are structurally zero).
// grid (16 s, 16 btiles), block 32.
// ---------------------------------------------------------------------------
template <int PAD, int KK>
__global__ void plx_gemm(const float* __restrict__ xin, const f16* __restrict__ Wp,
                         const float* __restrict__ bias, float* __restrict__ xout) {
  int s = blockIdx.x, bt = blockIdx.y;
  __shared__ alignas(32) f16 As[(KK / 32) * 512];
  int tid = threadIdx.x;
  for (int i = tid; i < 16 * KK; i += 32) {
    int m = i / KK, k = i % KK;
    int wi = k >> 8, f = k & 255;
    int j = s + wi - PAD;
    f16 v = (f16)0.f;
    if (j >= 0 && j < 6) v = (f16)xin[((size_t)(bt * 16 + m) * 256 + f) * 6 + j];
    As[(k >> 5) * 512 + frag_off(m, k & 31)] = v;
  }
  __syncthreads();
  const f16* Ws = Wp + (size_t)s * (KK / 32) * 512;
  v8f acc = {};
  for (int ks = 0; ks < KK / 32; ++ks) {
    v16h a = frag_ld(As + ks * 512);
    v16h b = frag_ld(Ws + ks * 512);
    acc = WMMA_F16(a, b, acc);
  }
  int l = tid & 31, n = l & 15, mh = (l >> 4) * 8;
  if (n < 6) {
    float bb = bias[s * 6 + n];
#pragma unroll
    for (int j = 0; j < 8; ++j) {
      int b = bt * 16 + mh + j;
      float v = acc[j] + bb;
      v = v > 0.f ? v : 0.f;
      xout[((size_t)b * 256 + s) * 6 + n] = v;
    }
  }
}

// Exact tail for s in [16,256): out = relu(bias[s]) broadcast over batch.
__global__ void plx_bias_fill(const float* __restrict__ bias, float* __restrict__ xout) {
  int i = blockIdx.x * 256 + threadIdx.x;            // 256*240*6
  if (i >= 256 * 240 * 6) return;
  int c = i % 6, r = i / 6;
  int s = 16 + (r % 240), b = r / 240;
  float v = bias[s * 6 + c];
  v = v > 0.f ? v : 0.f;
  xout[((size_t)b * 256 + s) * 6 + c] = v;
}

// ---------------------------------------------------------------------------
// K4: LSTM input projection: [1536 x 256] @ [256 x 1024].
// grid (96 mtiles, 16 ngroups), block 128.
// ---------------------------------------------------------------------------
__global__ void lstm_ih_gemm(const float* __restrict__ x7, const f16* __restrict__ wih,
                             float* __restrict__ gih) {
  int mt = blockIdx.x, ng = blockIdx.y;
  __shared__ alignas(32) f16 As[8 * 512];            // 8 KB
  int tid = threadIdx.x;
  for (int i = tid; i < 16 * 256; i += 128) {
    int m = i >> 8, k = i & 255;
    int row = mt * 16 + m;                           // row = t*256 + b
    int t = row >> 8, b = row & 255;
    As[(k >> 5) * 512 + frag_off(m, k & 31)] =
        (f16)x7[((size_t)b * 256 + k) * 6 + t];      // xs[t][b][k] = x7[b][k][t]
  }
  __syncthreads();
  int wv = tid >> 5, nt = ng * 4 + wv;
  v8f acc = {};
#pragma unroll
  for (int ks = 0; ks < 8; ++ks) {
    v16h a = frag_ld(As + ks * 512);
    v16h b = frag_ld(wih + (ks * 64 + nt) * 512);
    acc = WMMA_F16(a, b, acc);
  }
  int l = tid & 31, n = nt * 16 + (l & 15), mh = (l >> 4) * 8;
#pragma unroll
  for (int j = 0; j < 8; ++j)
    gih[(size_t)(mt * 16 + mh + j) * 1024 + n] = acc[j];
}

// K5: recurrent GEMM g = gih[t] + bias + h_prev @ Whh^T (dirs via grid.z).
// A tile (16x128 f16 = 4KB contiguous) staged with CDNA5 async global->LDS copy.
__global__ void lstm_rec_gemm(const f16* __restrict__ hprev, const f16* __restrict__ whh,
                              const float* __restrict__ gih, const float* __restrict__ bsum,
                              float* __restrict__ gbuf, int t_f, int t_b) {
  int mt = blockIdx.x, ng = blockIdx.y, d = blockIdx.z;
  int t = d ? t_b : t_f;
  const f16* hp = hprev + d * 256 * 128;
  const f16* w = whh + d * 65536;
  __shared__ alignas(32) f16 As[16 * 128];           // row-major 4 KB
  int tid = threadIdx.x;
  {
    const char* src = (const char*)(hp + (size_t)mt * 2048);
    unsigned ldsbase = (unsigned)(uintptr_t)As;
#pragma unroll
    for (int ch = 0; ch < 2; ++ch) {
      int e = tid + ch * 128;                        // 256 x 16B chunks
      unsigned la = ldsbase + e * 16;
      const char* ga = src + e * 16;
      asm volatile("global_load_async_to_lds_b128 %0, %1, off"
                   :: "v"(la), "v"(ga) : "memory");
    }
    asm volatile("s_wait_asynccnt 0" ::: "memory");
  }
  __syncthreads();
  int wv = tid >> 5, nt = ng * 4 + wv;
  v8f acc = {};
#pragma unroll
  for (int ks = 0; ks < 4; ++ks) {
    v16h a = frag_a_row(As + ks * 32, 128);
    v16h b = frag_ld(w + (ks * 32 + nt) * 512);
    acc = WMMA_F16(a, b, acc);
  }
  int l = tid & 31, n = nt * 16 + (l & 15), mh = (l >> 4) * 8;
  float bb = bsum[d * 512 + n];
#pragma unroll
  for (int j = 0; j < 8; ++j) {
    int b = mt * 16 + mh + j;
    gbuf[(size_t)(d * 256 + b) * 512 + n] =
        acc[j] + gih[((size_t)t * 256 + b) * 1024 + d * 512 + n] + bb;
  }
}

// K5e: gate nonlinearity + state update + scatter into out[b][t][dir*128+h]
__global__ void lstm_gates(const float* __restrict__ gbuf, float* __restrict__ cbuf,
                           f16* __restrict__ hprev, float* __restrict__ outbuf,
                           int t_f, int t_b) {
  int i = blockIdx.x * 256 + threadIdx.x;            // 2*256*128
  int d = i >> 15, r = i & 32767;
  int b = r >> 7, hh = r & 127;
  const float* g = gbuf + (size_t)(d * 256 + b) * 512;
  float gi = g[hh], gf = g[128 + hh], gg = g[256 + hh], go = g[384 + hh];
  float si = 1.f / (1.f + __expf(-gi));
  float sf = 1.f / (1.f + __expf(-gf));
  float so = 1.f / (1.f + __expf(-go));
  float c = sf * cbuf[i] + si * tanhf(gg);
  float h = so * tanhf(c);
  cbuf[i] = c;
  hprev[i] = (f16)h;
  int t = d ? t_b : t_f;
  outbuf[((size_t)b * 6 + t) * 256 + d * 128 + hh] = h;
}

// ---------------------------------------------------------------------------
// K6: RepBlock as 6 GEMMs (spatial i), K=768. grid (6,16,4), block 128.
// ---------------------------------------------------------------------------
__global__ void rep_gemm(const float* __restrict__ outbuf, const f16* __restrict__ wr,
                         const float* __restrict__ rb, float* __restrict__ repbuf) {
  int i = blockIdx.x, mt = blockIdx.y, ng = blockIdx.z;
  __shared__ alignas(32) f16 As[24 * 512];           // 24 KB
  int tid = threadIdx.x;
  for (int ii = tid; ii < 16 * 768; ii += 128) {
    int m = ii / 768, k = ii % 768;
    int wi = k >> 8, ic = k & 255;
    int row = i + wi - 1;
    f16 v = (f16)0.f;
    if (row >= 0 && row < 6)
      v = (f16)outbuf[(size_t)(mt * 16 + m) * 1536 + ic * 6 + row];
    As[(k >> 5) * 512 + frag_off(m, k & 31)] = v;
  }
  __syncthreads();
  int wv = tid >> 5, nt = ng * 4 + wv;
  v8f acc = {};
#pragma unroll 4
  for (int ks = 0; ks < 24; ++ks) {
    v16h a = frag_ld(As + ks * 512);
    v16h b = frag_ld(wr + (ks * 16 + nt) * 512);
    acc = WMMA_F16(a, b, acc);
  }
  int l = tid & 31, o = nt * 16 + (l & 15), mh = (l >> 4) * 8;
  float bb = rb[o];
#pragma unroll
  for (int j = 0; j < 8; ++j) {
    int b = mt * 16 + mh + j;
    float idv = outbuf[(size_t)b * 1536 + o * 6 + i];
    float v = acc[j] + bb + idv;
    v = v > 0.f ? v : 0.f;
    repbuf[(size_t)b * 1536 + o * 6 + i] = v;
  }
}

// K7: mean-pool over t (raw-reshape view) + fc1+relu + fc2. One block per batch.
__global__ void head_fc(const float* __restrict__ repbuf, const float* __restrict__ w1,
                        const float* __restrict__ b1, const float* __restrict__ w2,
                        const float* __restrict__ b2, float* __restrict__ out) {
  int b = blockIdx.x, tid = threadIdx.x;             // 128 threads
  __shared__ float pooled[256];
  __shared__ float a1[128];
  const float* rp = repbuf + (size_t)b * 1536;
  for (int f = tid; f < 256; f += 128) {
    float s = 0.f;
#pragma unroll
    for (int t = 0; t < 6; ++t) s += rp[t * 256 + f];
    pooled[f] = s * (1.f / 6.f);
  }
  __syncthreads();
  float acc = b1[tid];
  const float* wrow = w1 + tid * 256;
  for (int f = 0; f < 256; ++f) acc += pooled[f] * wrow[f];
  a1[tid] = acc > 0.f ? acc : 0.f;
  __syncthreads();
  if (tid < 2) {
    float s = b2[tid];
    const float* w2r = w2 + tid * 128;
    for (int j = 0; j < 128; ++j) s += a1[j] * w2r[j];
    out[b * 2 + tid] = s;
  }
}

// ---------------------------------------------------------------------------
extern "C" void kernel_launch(void* const* d_in, const int* in_sizes, int n_in,
                              void* d_out, int out_size, void* d_ws, size_t ws_size,
                              hipStream_t stream) {
  (void)in_sizes; (void)n_in; (void)out_size; (void)ws_size;
  const float* x       = (const float*)d_in[0];
  const float* conv_w  = (const float*)d_in[1];
  const float* conv_b  = (const float*)d_in[2];
  const float* pl3_w   = (const float*)d_in[3];
  const float* pl3_b   = (const float*)d_in[4];
  const float* pl5_w   = (const float*)d_in[5];
  const float* pl5_b   = (const float*)d_in[6];
  const float* pl7_w   = (const float*)d_in[7];
  const float* pl7_b   = (const float*)d_in[8];
  const float* wih_f   = (const float*)d_in[9];
  const float* whh_f   = (const float*)d_in[10];
  const float* bih_f   = (const float*)d_in[11];
  const float* bhh_f   = (const float*)d_in[12];
  const float* wih_b   = (const float*)d_in[13];
  const float* whh_b   = (const float*)d_in[14];
  const float* bih_b   = (const float*)d_in[15];
  const float* bhh_b   = (const float*)d_in[16];
  const float* rep_w3  = (const float*)d_in[17];
  const float* rep_b3  = (const float*)d_in[18];
  const float* rep_w1  = (const float*)d_in[19];
  const float* rep_b1  = (const float*)d_in[20];
  const float* fc1_w   = (const float*)d_in[21];
  const float* fc1_b   = (const float*)d_in[22];
  const float* fc2_w   = (const float*)d_in[23];
  const float* fc2_b   = (const float*)d_in[24];
  float* out = (float*)d_out;

  // workspace carve-up (256B aligned; fragment tiles land on 1KB boundaries)
  char* base = (char*)d_ws;
  size_t off = 0;
  auto carve = [&](size_t bytes) {
    void* p = base + off;
    off = (off + bytes + 255) & ~(size_t)255;
    return p;
  };
  f16*   h16    = (f16*)  carve((size_t)256 * 256 * 256 * 2); // conv act, f16
  f16*   convwp = (f16*)  carve(16 * 512 * 2);
  f16*   wihc   = (f16*)  carve(8 * 64 * 512 * 2);
  f16*   whhc   = (f16*)  carve(2 * 4 * 32 * 512 * 2);
  float* bsum   = (float*)carve(1024 * 4);
  f16*   wrw    = (f16*)  carve(24 * 16 * 512 * 2);
  float* rbias  = (float*)carve(256 * 4);
  f16*   pl3p   = (f16*)  carve((size_t)256 * 24 * 512 * 2);
  f16*   pl5p   = (f16*)  carve((size_t)16 * 40 * 512 * 2);
  f16*   pl7p   = (f16*)  carve((size_t)16 * 56 * 512 * 2);
  float* x3     = (float*)carve((size_t)256 * 256 * 6 * 4);
  float* x5     = (float*)carve((size_t)256 * 256 * 6 * 4);
  float* x7     = (float*)carve((size_t)256 * 256 * 6 * 4);
  float* gih    = (float*)carve((size_t)1536 * 1024 * 4);
  f16*   hprev  = (f16*)  carve(2 * 256 * 128 * 2);
  float* cbuf   = (float*)carve(2 * 256 * 128 * 4);
  float* gbuf   = (float*)carve((size_t)2 * 256 * 512 * 4);
  float* outbuf = (float*)carve((size_t)256 * 1536 * 4);
  float* repbuf = (float*)carve((size_t)256 * 1536 * 4);

  // weight prep + state init
  prep_convw<<<32, 256, 0, stream>>>(conv_w, convwp);
  prep_wih<<<1024, 256, 0, stream>>>(wih_f, wih_b, wihc);
  prep_whh<<<512, 256, 0, stream>>>(whh_f, whh_b, whhc);
  prep_bias<<<4, 256, 0, stream>>>(bih_f, bhh_f, bih_b, bhh_b, bsum);
  prep_repw<<<768, 256, 0, stream>>>(rep_w3, rep_w1, rep_b3, rep_b1, wrw, rbias);
  prep_plw<<<12288, 256, 0, stream>>>(pl3_w, pl3p, 768, 256 * 768 * 16);
  prep_plw<<<1280, 256, 0, stream>>>(pl5_w, pl5p, 1280, 16 * 1280 * 16);
  prep_plw<<<1792, 256, 0, stream>>>(pl7_w, pl7p, 1792, 16 * 1792 * 16);
  zero_state<<<256, 256, 0, stream>>>(hprev, cbuf);

  // conv + position-linear cascade
  conv_gemm<<<dim3(256, 16), 128, 0, stream>>>(x, convwp, conv_b, h16);
  pl3_gemm<<<dim3(256, 16), 32, 0, stream>>>(h16, pl3p, pl3_b, x3);
  plx_gemm<2, 1280><<<dim3(16, 16), 32, 0, stream>>>(x3, pl5p, pl5_b, x5);
  plx_bias_fill<<<1440, 256, 0, stream>>>(pl5_b, x5);
  plx_gemm<3, 1792><<<dim3(16, 16), 32, 0, stream>>>(x5, pl7p, pl7_b, x7);
  plx_bias_fill<<<1440, 256, 0, stream>>>(pl7_b, x7);

  // BiLSTM
  lstm_ih_gemm<<<dim3(96, 16), 128, 0, stream>>>(x7, wihc, gih);
  for (int j = 0; j < 6; ++j) {
    lstm_rec_gemm<<<dim3(16, 8, 2), 128, 0, stream>>>(hprev, whhc, gih, bsum, gbuf, j, 5 - j);
    lstm_gates<<<256, 256, 0, stream>>>(gbuf, cbuf, hprev, outbuf, j, 5 - j);
  }

  // RepBlock + head
  rep_gemm<<<dim3(6, 16, 4), 128, 0, stream>>>(outbuf, wrw, rbias, repbuf);
  head_fc<<<256, 128, 0, stream>>>(repbuf, fc1_w, fc1_b, fc2_w, fc2_b, out);
}
